// LSTMTagger_49959059587773
// MI455X (gfx1250) — matrix operand order repaired and checked by reading, not compiled, not run on hardware
//
#include <hip/hip_runtime.h>
#include <hip/hip_bf16.h>
#include <math.h>

// ---------------- problem constants ----------------
constexpr int S_    = 512;
constexpr int BATCH = 32;
constexpr int E_    = 256;
constexpr int H_    = 256;
constexpr int G4    = 4 * H_;          // 1024
constexpr int TAGS  = 64;
constexpr int MROWS = S_ * BATCH;      // 16384 (time-major row index t*32+b)

typedef __bf16 bf16_t;
typedef __attribute__((ext_vector_type(16))) __bf16 v16bf;
typedef __attribute__((ext_vector_type(8)))  float  v8f;

// ---------------- WMMA fragment helpers (wave32, 16x16x32 bf16) ----------------
// A fragment: 16x32 (MxK), row-major source with leading dim lda (bf16).
// Layout (ISA 7.12.2): lanes 0-15 -> M=0..15, j=0..7 => K=half*8+j, j=8..15 => K=16+half*8+(j-8)
__device__ __forceinline__ v16bf load_a_frag(const bf16_t* __restrict__ A, int lda) {
    int lane = threadIdx.x & 31;
    int m    = lane & 15;
    int h8   = (lane >> 4) * 8;
    const bf16_t* p = A + m * lda;
    v16bf a;
#pragma unroll
    for (int j = 0; j < 8; ++j) a[j]     = p[h8 + j];
#pragma unroll
    for (int j = 0; j < 8; ++j) a[8 + j] = p[16 + h8 + j];
    return a;
}

// A fragment built from the sum of two fp32 sources (enc = hf + hb), converted to bf16.
__device__ __forceinline__ v16bf load_a_frag_sum(const float* __restrict__ A0,
                                                 const float* __restrict__ A1, int lda) {
    int lane = threadIdx.x & 31;
    int m    = lane & 15;
    int h8   = (lane >> 4) * 8;
    const float* p0 = A0 + m * lda;
    const float* p1 = A1 + m * lda;
    v16bf a;
#pragma unroll
    for (int j = 0; j < 8; ++j) a[j]     = (bf16_t)(p0[h8 + j]      + p1[h8 + j]);
#pragma unroll
    for (int j = 0; j < 8; ++j) a[8 + j] = (bf16_t)(p0[16 + h8 + j] + p1[16 + h8 + j]);
    return a;
}

// B fragment: 32x16 (KxN) taken from a TRANSPOSED weight matrix WT[N][K] (bf16),
// tile origin passed as WT + n0*ldk + k0. Lane n = lane&15, j=0..7 => K = half*8+j, j=8..15 => K = 16+half*8+(j-8).
__device__ __forceinline__ v16bf load_b_frag(const bf16_t* __restrict__ WT, int ldk) {
    int lane = threadIdx.x & 31;
    int n    = lane & 15;
    int h8   = (lane >> 4) * 8;
    const bf16_t* p = WT + n * ldk;
    v16bf b;
#pragma unroll
    for (int j = 0; j < 8; ++j) b[j]     = p[h8 + j];
#pragma unroll
    for (int j = 0; j < 8; ++j) b[8 + j] = p[16 + h8 + j];
    return b;
}

// C/D fragment: 16x16 f32. lane n = lane&15, VGPR r -> row half*8+r.
__device__ __forceinline__ v8f load_c_frag(const float* __restrict__ C, int ldc) {
    int lane = threadIdx.x & 31;
    int n    = lane & 15;
    int m0   = (lane >> 4) * 8;
    v8f c;
#pragma unroll
    for (int r = 0; r < 8; ++r) c[r] = C[(m0 + r) * ldc + n];
    return c;
}

__device__ __forceinline__ void store_c_frag(float* __restrict__ C, int ldc, v8f c) {
    int lane = threadIdx.x & 31;
    int n    = lane & 15;
    int m0   = (lane >> 4) * 8;
#pragma unroll
    for (int r = 0; r < 8; ++r) C[(m0 + r) * ldc + n] = c[r];
}

__device__ __forceinline__ v8f wmma_bf16(v16bf a, v16bf b, v8f c) {
    return __builtin_amdgcn_wmma_f32_16x16x32_bf16(false, a, false, b, (short)0, c,
                                                   false, false);
}

__device__ __forceinline__ float sigf(float x) { return 1.0f / (1.0f + __expf(-x)); }

// ---------------- kernel 1: convert weight fp32 [K][N] -> transposed bf16 [N][K] -----
__global__ void k_cvt_wT(const float* __restrict__ W, bf16_t* __restrict__ WT,
                         int K, int N) {
    int idx = blockIdx.x * blockDim.x + threadIdx.x;
    if (idx >= K * N) return;
    int n = idx / K;
    int k = idx - n * K;
    WT[idx] = (bf16_t)W[k * N + n];
}

// ---------------- kernel 2: embedding gather -> time-major bf16 X[S][B][E] ---------
__global__ void k_embed(const int* __restrict__ sent, const float* __restrict__ emb,
                        bf16_t* __restrict__ X) {
    int idx = blockIdx.x * blockDim.x + threadIdx.x;     // over S*B*E
    if (idx >= S_ * BATCH * E_) return;
    int e  = idx & (E_ - 1);
    int b  = (idx >> 8) & (BATCH - 1);
    int t  = idx >> 13;
    int tok = sent[b * S_ + t];
    X[idx] = (bf16_t)emb[tok * E_ + e];
}

// ---------------- kernel 3: generic big GEMM  C[M][N] = A[M][K] @ WT^T --------------
__global__ void k_gemm_bf16(const bf16_t* __restrict__ A, const bf16_t* __restrict__ WT,
                            float* __restrict__ C, int Mt, int Nt, int K) {
    int wave = (int)(threadIdx.x >> 5);
    int tile = blockIdx.x * (blockDim.x >> 5) + wave;
    if (tile >= Mt * Nt) return;
    int mt = tile / Nt, nt = tile - mt * Nt;
    int N  = Nt * 16;
    const bf16_t* Ap = A  + (size_t)mt * 16 * K;
    const bf16_t* Bp = WT + (size_t)nt * 16 * K;
    v8f acc = {};
    for (int k0 = 0; k0 < K; k0 += 32) {
        v16bf a = load_a_frag(Ap + k0, K);
        v16bf b = load_b_frag(Bp + k0, K);
        acc = wmma_bf16(a, b, acc);
    }
    store_c_frag(C + (size_t)mt * 16 * N + nt * 16, N, acc);
}

// ---------------- kernel 4: gdx = (hf+hb) @ W1   (A summed on the fly) --------------
__global__ void k_gemm_enc(const float* __restrict__ HF, const float* __restrict__ HB,
                           const bf16_t* __restrict__ WT, float* __restrict__ C,
                           int Mt, int Nt, int K) {
    int wave = (int)(threadIdx.x >> 5);
    int tile = blockIdx.x * (blockDim.x >> 5) + wave;
    if (tile >= Mt * Nt) return;
    int mt = tile / Nt, nt = tile - mt * Nt;
    int N  = Nt * 16;
    const float*  A0 = HF + (size_t)mt * 16 * K;
    const float*  A1 = HB + (size_t)mt * 16 * K;
    const bf16_t* Bp = WT + (size_t)nt * 16 * K;
    v8f acc = {};
    for (int k0 = 0; k0 < K; k0 += 32) {
        v16bf a = load_a_frag_sum(A0 + k0, A1 + k0, K);
        v16bf b = load_b_frag(Bp + k0, K);
        acc = wmma_bf16(a, b, acc);
    }
    store_c_frag(C + (size_t)mt * 16 * N + nt * 16, N, acc);
}

// ---------------- kernel 5: persistent bidirectional encoder ------------------------
// grid = 2 blocks (dir 0 fwd / 1 bwd), 1024 threads = 32 waves on one WGP.
// State: c fp32 in registers (fixed element ownership), h/c as bf16 in LDS (WMMA A).
__global__ __launch_bounds__(1024)
void k_encoder(const float* __restrict__ gfx, const float* __restrict__ gbx,
               const bf16_t* __restrict__ wfhT, const bf16_t* __restrict__ wfcT,
               const bf16_t* __restrict__ wbhT, const bf16_t* __restrict__ wbcT,
               float* __restrict__ hf, float* __restrict__ hb,
               float* __restrict__ gscr) {
    __shared__ bf16_t h_bf[BATCH * H_];
    __shared__ bf16_t c_bf[BATCH * H_];

    const int dir  = blockIdx.x;
    const float*  gx  = dir ? gbx  : gfx;
    const bf16_t* WhT = dir ? wbhT : wfhT;
    const bf16_t* WcT = dir ? wbcT : wfcT;
    float* hout = dir ? hb : hf;
    float* gs   = gscr + (size_t)dir * BATCH * G4;

    const int tid  = threadIdx.x;
    const int wave = tid >> 5;

    float creg[8];
#pragma unroll
    for (int r = 0; r < 8; ++r) {
        creg[r] = 0.0f;
        int e = tid * 8 + r;
        h_bf[e] = (bf16_t)0.0f;
        c_bf[e] = (bf16_t)0.0f;
    }
    __syncthreads();

#pragma unroll 1
    for (int s = 0; s < S_; ++s) {
        const int t = dir ? (S_ - 1 - s) : s;
        const float* gxt = gx + (size_t)t * BATCH * G4;

        // prefetch next timestep's x-contribution into cache hierarchy
        if (s + 1 < S_) {
            const int tn = dir ? (t - 1) : (t + 1);
            __builtin_prefetch(gx + (size_t)tn * BATCH * G4 + tid * 32, 0, 1);
        }

        // ---- phase 1: g = gx[t] + h@Wh + c@Wc  (128 tiles, 4 per wave) ----
#pragma unroll
        for (int i = 0; i < 4; ++i) {
            int tile = wave * 4 + i;           // 0..127
            int mt = tile >> 6;                // 0..1
            int nt = tile & 63;                // 0..63
            v8f acc = load_c_frag(gxt + mt * 16 * G4 + nt * 16, G4);
            const bf16_t* bh = WhT + (size_t)nt * 16 * H_;
            const bf16_t* bc = WcT + (size_t)nt * 16 * H_;
            for (int k0 = 0; k0 < H_; k0 += 32) {
                v16bf ah = load_a_frag(h_bf + mt * 16 * H_ + k0, H_);
                v16bf bbh = load_b_frag(bh + k0, H_);
                acc = wmma_bf16(ah, bbh, acc);
                v16bf ac = load_a_frag(c_bf + mt * 16 * H_ + k0, H_);
                v16bf bbc = load_b_frag(bc + k0, H_);
                acc = wmma_bf16(ac, bbc, acc);
            }
            store_c_frag(gs + mt * 16 * G4 + nt * 16, G4, acc);
        }
        __threadfence();
        __syncthreads();

        // ---- phase 2: pointwise LSTM update ----
#pragma unroll
        for (int r = 0; r < 8; ++r) {
            int e = tid * 8 + r;
            int b = e >> 8;
            int j = e & (H_ - 1);
            const float* gb = gs + b * G4;
            float gi = gb[j], gf = gb[H_ + j], gg = gb[2 * H_ + j], go = gb[3 * H_ + j];
            float cn = sigf(gf) * creg[r] + sigf(gi) * tanhf(gg);
            float hn = sigf(go) * tanhf(cn);
            creg[r] = cn;
            h_bf[e] = (bf16_t)hn;
            c_bf[e] = (bf16_t)cn;
            hout[(size_t)t * BATCH * H_ + e] = hn;
        }
        __syncthreads();
    }
}

// ---------------- kernel 6: persistent decoder (3-state cell) -----------------------
__global__ __launch_bounds__(1024)
void k_decoder(const float* __restrict__ gdx,
               const bf16_t* __restrict__ w2T, const bf16_t* __restrict__ w3T,
               const bf16_t* __restrict__ w4T, const bf16_t* __restrict__ w5T,
               bf16_t* __restrict__ decbf, float* __restrict__ gs) {
    __shared__ bf16_t h_bf[BATCH * H_];
    __shared__ bf16_t c_bf[BATCH * H_];
    __shared__ bf16_t m_bf[BATCH * H_];

    const int tid  = threadIdx.x;
    const int wave = tid >> 5;

    float creg[8];
#pragma unroll
    for (int r = 0; r < 8; ++r) {
        creg[r] = 0.0f;
        int e = tid * 8 + r;
        h_bf[e] = (bf16_t)0.0f;
        c_bf[e] = (bf16_t)0.0f;
        m_bf[e] = (bf16_t)0.0f;
    }
    __syncthreads();

#pragma unroll 1
    for (int t = 0; t < S_; ++t) {
        const float* gxt = gdx + (size_t)t * BATCH * G4;
        if (t + 1 < S_)
            __builtin_prefetch(gdx + (size_t)(t + 1) * BATCH * G4 + tid * 32, 0, 1);

        // ---- phase 1: g = gx[t] + h@W2 + c@W3 + m@W4 ----
#pragma unroll
        for (int i = 0; i < 4; ++i) {
            int tile = wave * 4 + i;
            int mt = tile >> 6;
            int nt = tile & 63;
            v8f acc = load_c_frag(gxt + mt * 16 * G4 + nt * 16, G4);
            const bf16_t* b2 = w2T + (size_t)nt * 16 * H_;
            const bf16_t* b3 = w3T + (size_t)nt * 16 * H_;
            const bf16_t* b4 = w4T + (size_t)nt * 16 * H_;
            for (int k0 = 0; k0 < H_; k0 += 32) {
                acc = wmma_bf16(load_a_frag(h_bf + mt * 16 * H_ + k0, H_),
                                load_b_frag(b2 + k0, H_), acc);
                acc = wmma_bf16(load_a_frag(c_bf + mt * 16 * H_ + k0, H_),
                                load_b_frag(b3 + k0, H_), acc);
                acc = wmma_bf16(load_a_frag(m_bf + mt * 16 * H_ + k0, H_),
                                load_b_frag(b4 + k0, H_), acc);
            }
            store_c_frag(gs + mt * 16 * G4 + nt * 16, G4, acc);
        }
        __threadfence();
        __syncthreads();

        // ---- phase 2: pointwise update of h, c ----
#pragma unroll
        for (int r = 0; r < 8; ++r) {
            int e = tid * 8 + r;
            int b = e >> 8;
            int j = e & (H_ - 1);
            const float* gb = gs + b * G4;
            float gi = gb[j], gf = gb[H_ + j], gg = gb[2 * H_ + j], go = gb[3 * H_ + j];
            float cn = sigf(gf) * creg[r] + sigf(gi) * tanhf(gg);
            float hn = sigf(go) * tanhf(cn);
            creg[r] = cn;
            bf16_t hb16 = (bf16_t)hn;
            h_bf[e] = hb16;
            c_bf[e] = (bf16_t)cn;
            decbf[(size_t)t * BATCH * H_ + e] = hb16;   // decoder output = h
        }
        __syncthreads();

        // ---- phase 3: m = tanh(h @ W5)  (32 tiles, 1 per wave) ----
        {
            int mt = wave >> 4;                // 0..1
            int nt = wave & 15;                // 0..15
            v8f acc = {};
            const bf16_t* b5 = w5T + (size_t)nt * 16 * H_;
            for (int k0 = 0; k0 < H_; k0 += 32) {
                acc = wmma_bf16(load_a_frag(h_bf + mt * 16 * H_ + k0, H_),
                                load_b_frag(b5 + k0, H_), acc);
            }
            int lane = threadIdx.x & 31;
            int n  = lane & 15;
            int m0 = (lane >> 4) * 8;
#pragma unroll
            for (int r = 0; r < 8; ++r)
                m_bf[(mt * 16 + m0 + r) * H_ + nt * 16 + n] = (bf16_t)tanhf(acc[r]);
        }
        __syncthreads();
    }
}

// ---------------- kernel 7: logits = dec @ out_w + b  (WMMA) ------------------------
__global__ void k_logits(const bf16_t* __restrict__ dec, const bf16_t* __restrict__ woutT,
                         const float* __restrict__ out_b, float* __restrict__ logits) {
    int wave = (int)(threadIdx.x >> 5);
    int tile = blockIdx.x * (blockDim.x >> 5) + wave;
    const int Nt = TAGS / 16;                 // 4
    if (tile >= (MROWS / 16) * Nt) return;
    int mt = tile / Nt, nt = tile - mt * Nt;
    int lane = threadIdx.x & 31;
    int n = lane & 15;
    float bias = out_b[nt * 16 + n];
    v8f acc;
#pragma unroll
    for (int r = 0; r < 8; ++r) acc[r] = bias;
    const bf16_t* Ap = dec   + (size_t)mt * 16 * H_;
    const bf16_t* Bp = woutT + (size_t)nt * 16 * H_;
    for (int k0 = 0; k0 < H_; k0 += 32) {
        acc = wmma_bf16(load_a_frag(Ap + k0, H_), load_b_frag(Bp + k0, H_), acc);
    }
    store_c_frag(logits + (size_t)mt * 16 * TAGS + nt * 16, TAGS, acc);
}

// ---------------- kernel 8: log_softmax + [S,B] -> [B,S] transpose ------------------
__global__ void k_logsoftmax(const float* __restrict__ logits, float* __restrict__ out) {
    int row = blockIdx.x * blockDim.x + threadIdx.x;
    if (row >= MROWS) return;
    const float* p = logits + (size_t)row * TAGS;
    float mx = -INFINITY;
#pragma unroll
    for (int k = 0; k < TAGS; ++k) mx = fmaxf(mx, p[k]);
    float s = 0.0f;
#pragma unroll
    for (int k = 0; k < TAGS; ++k) s += __expf(p[k] - mx);
    float lse = mx + __logf(s);
    int t = row >> 5, b = row & 31;
    float* q = out + ((size_t)b * S_ + t) * TAGS;
#pragma unroll
    for (int k = 0; k < TAGS; ++k) q[k] = p[k] - lse;
}

// ---------------- host side ---------------------------------------------------------
extern "C" void kernel_launch(void* const* d_in, const int* in_sizes, int n_in,
                              void* d_out, int out_size, void* d_ws, size_t ws_size,
                              hipStream_t stream) {
    const int*   sent   = (const int*)  d_in[0];
    const float* emb    = (const float*)d_in[1];
    const float* enc_fx = (const float*)d_in[2];
    const float* enc_fh = (const float*)d_in[3];
    const float* enc_fc = (const float*)d_in[4];
    const float* enc_bx = (const float*)d_in[5];
    const float* enc_bh = (const float*)d_in[6];
    const float* enc_bc = (const float*)d_in[7];
    const float* dec_w1 = (const float*)d_in[8];
    const float* dec_w2 = (const float*)d_in[9];
    const float* dec_w3 = (const float*)d_in[10];
    const float* dec_w4 = (const float*)d_in[11];
    const float* dec_w5 = (const float*)d_in[12];
    const float* out_w  = (const float*)d_in[13];
    const float* out_b  = (const float*)d_in[14];
    float* out = (float*)d_out;

    // ---- workspace layout (bytes) ----
    char* ws = (char*)d_ws;
    size_t off = 0;
    auto alloc = [&](size_t bytes) { char* p = ws + off; off += (bytes + 255) & ~size_t(255); return p; };

    bf16_t* Xbf   = (bf16_t*)alloc((size_t)MROWS * E_ * 2);
    bf16_t* wfxT  = (bf16_t*)alloc((size_t)E_ * G4 * 2);
    bf16_t* wfhT  = (bf16_t*)alloc((size_t)H_ * G4 * 2);
    bf16_t* wfcT  = (bf16_t*)alloc((size_t)H_ * G4 * 2);
    bf16_t* wbxT  = (bf16_t*)alloc((size_t)E_ * G4 * 2);
    bf16_t* wbhT  = (bf16_t*)alloc((size_t)H_ * G4 * 2);
    bf16_t* wbcT  = (bf16_t*)alloc((size_t)H_ * G4 * 2);
    bf16_t* w1T   = (bf16_t*)alloc((size_t)H_ * G4 * 2);
    bf16_t* w2T   = (bf16_t*)alloc((size_t)H_ * G4 * 2);
    bf16_t* w3T   = (bf16_t*)alloc((size_t)H_ * G4 * 2);
    bf16_t* w4T   = (bf16_t*)alloc((size_t)H_ * G4 * 2);
    bf16_t* w5T   = (bf16_t*)alloc((size_t)H_ * H_ * 2);
    bf16_t* woutT = (bf16_t*)alloc((size_t)H_ * TAGS * 2);
    float*  gfx   = (float*)alloc((size_t)MROWS * G4 * 4);
    float*  gbx   = (float*)alloc((size_t)MROWS * G4 * 4);   // later aliased as gdx
    float*  hf    = (float*)alloc((size_t)MROWS * H_ * 4);
    float*  hb    = (float*)alloc((size_t)MROWS * H_ * 4);
    bf16_t* decbf = (bf16_t*)alloc((size_t)MROWS * H_ * 2);
    float*  logit = (float*)alloc((size_t)MROWS * TAGS * 4);
    float*  gscr  = (float*)alloc((size_t)2 * BATCH * G4 * 4);
    float*  gdx   = gbx;   // reuse after backward encoder finishes (stream-ordered)

    // ---- 1. weight conversions (fp32 [K][N] -> bf16 transposed [N][K]) ----
    auto cvt = [&](const float* w, bf16_t* wt, int K, int N) {
        int tot = K * N;
        k_cvt_wT<<<(tot + 255) / 256, 256, 0, stream>>>(w, wt, K, N);
    };
    cvt(enc_fx, wfxT, E_, G4);  cvt(enc_fh, wfhT, H_, G4);  cvt(enc_fc, wfcT, H_, G4);
    cvt(enc_bx, wbxT, E_, G4);  cvt(enc_bh, wbhT, H_, G4);  cvt(enc_bc, wbcT, H_, G4);
    cvt(dec_w1, w1T, H_, G4);   cvt(dec_w2, w2T, H_, G4);
    cvt(dec_w3, w3T, H_, G4);   cvt(dec_w4, w4T, H_, G4);
    cvt(dec_w5, w5T, H_, H_);   cvt(out_w, woutT, H_, TAGS);

    // ---- 2. embedding gather ----
    {
        int tot = MROWS * E_;
        k_embed<<<(tot + 255) / 256, 256, 0, stream>>>(sent, emb, Xbf);
    }

    // ---- 3. time-parallel x-contributions: gfx = X@Wfx, gbx = X@Wbx ----
    {
        int Mt = MROWS / 16, Nt = G4 / 16;
        int blocks = (Mt * Nt + 7) / 8;
        k_gemm_bf16<<<blocks, 256, 0, stream>>>(Xbf, wfxT, gfx, Mt, Nt, E_);
        k_gemm_bf16<<<blocks, 256, 0, stream>>>(Xbf, wbxT, gbx, Mt, Nt, E_);
    }

    // ---- 4. bidirectional recurrent encoder (persistent, 2 WGPs) ----
    k_encoder<<<2, 1024, 0, stream>>>(gfx, gbx, wfhT, wfcT, wbhT, wbcT, hf, hb, gscr);

    // ---- 5. decoder x-contribution: gdx = (hf+hb)@W1 (fused sum in A loader) ----
    {
        int Mt = MROWS / 16, Nt = G4 / 16;
        int blocks = (Mt * Nt + 7) / 8;
        k_gemm_enc<<<blocks, 256, 0, stream>>>(hf, hb, w1T, gdx, Mt, Nt, H_);
    }

    // ---- 6. recurrent decoder (persistent, 1 WGP) ----
    k_decoder<<<1, 1024, 0, stream>>>(gdx, w2T, w3T, w4T, w5T, decbf, gscr);

    // ---- 7. output projection + bias (WMMA) ----
    {
        int tiles = (MROWS / 16) * (TAGS / 16);
        k_logits<<<(tiles + 7) / 8, 256, 0, stream>>>(decbf, woutT, out_b, logit);
    }

    // ---- 8. log_softmax + transpose to [B,S,TAGS] ----
    k_logsoftmax<<<(MROWS + 255) / 256, 256, 0, stream>>>(logit, out);
}
